// Decoder_44117904065271
// MI455X (gfx1250) — matrix operand order repaired and checked by reading, not compiled, not run on hardware
//
#include <hip/hip_runtime.h>
#include <math.h>

// ---------------- problem constants ----------------
#define DIMQ   128   // output dim
#define HIDQ   256   // hidden dim
#define MAXNQ  64
#define BQ     8192
#define KMQ    160   // key-net mid
#define DMQ    192   // decoder mid
#define SMQ    128   // size-pred mid
#define LN_EPS 1e-5f

#define DEC_WAVES  16          // waves per decoder block
#define DEC_THREADS (DEC_WAVES * 32)
#define DEC_ROWS   (DEC_WAVES * 16)   // 256 rows per block

typedef __attribute__((ext_vector_type(16))) _Float16 v16h;
typedef __attribute__((ext_vector_type(8)))  float    v8f;

// ---- native CDNA5 tanh (V_TANH_F32); branch-free fallback ----
__device__ __forceinline__ float fast_tanh(float x) {
#if __has_builtin(__builtin_amdgcn_tanhf)
    return __builtin_amdgcn_tanhf(x);
#elif __has_builtin(__builtin_amdgcn_tanh_f32)
    return __builtin_amdgcn_tanh_f32(x);
#else
    // tanh(x) = 1 - 2/(exp2(2*log2e*x)+1); clamp keeps exp2 finite, no branches
    const float xc = fminf(fmaxf(x, -10.0f), 10.0f);
    const float e  = __builtin_amdgcn_exp2f(xc * 2.8853900817779268f);
    return 1.0f - 2.0f * __builtin_amdgcn_rcpf(e + 1.0f);
#endif
}

// ---- fragment loaders (LDS, 16B-aligned) ----
// B-matrix 32x16 f16 (wave32): per lane halves[0..7]=K base..base+7, halves[8..15]=K base+16..base+23;
// base = 0 for lane<16, 8 for lane>=16. Caller pre-offsets for hi half.
__device__ __forceinline__ v16h loadB16(const _Float16* p) {
    union { uint4 u[2]; v16h h; } x;
    x.u[0] = *(const uint4*)(p);
    x.u[1] = *(const uint4*)(p + 16);
    return x.h;
}
// A-matrix 16x32 f16: per lane halves[0..7]=K base..base+7, halves[8..15]=K base+16..base+23
__device__ __forceinline__ v16h loadA16(const _Float16* p) {
    union { uint4 u[2]; v16h h; } x;
    x.u[0] = *(const uint4*)(p);
    x.u[1] = *(const uint4*)(p + 16);
    return x.h;
}

// =====================================================================
// Kernel 1: keys = tanh(LN(kW1 + kb1)) @ kW2 + kb2   (pos is identity)
// =====================================================================
__global__ void __launch_bounds__(256) keys_kernel(
    const float* __restrict__ kW1, const float* __restrict__ kb1,
    const float* __restrict__ kg,  const float* __restrict__ kbe,
    const float* __restrict__ kW2, const float* __restrict__ kb2,
    float* __restrict__ keys)
{
    __shared__ float s_act[KMQ];
    __shared__ float s_red[256];
    const int j = blockIdx.x, t = threadIdx.x;

    float v = (t < KMQ) ? (kW1[j * KMQ + t] + kb1[t]) : 0.0f;
    s_red[t] = (t < KMQ) ? v : 0.0f; __syncthreads();
    for (int off = 128; off > 0; off >>= 1) { if (t < off) s_red[t] += s_red[t + off]; __syncthreads(); }
    const float mu = s_red[0] * (1.0f / KMQ); __syncthreads();
    const float d = v - mu;
    s_red[t] = (t < KMQ) ? d * d : 0.0f; __syncthreads();
    for (int off = 128; off > 0; off >>= 1) { if (t < off) s_red[t] += s_red[t + off]; __syncthreads(); }
    const float var = s_red[0] * (1.0f / KMQ); __syncthreads();
    if (t < KMQ) s_act[t] = fast_tanh(d * rsqrtf(var + LN_EPS) * kg[t] + kbe[t]);
    __syncthreads();

    float acc = kb2[t];
    #pragma unroll 4
    for (int k = 0; k < KMQ; ++k) acc += s_act[k] * kW2[k * HIDQ + t];
    keys[j * HIDQ + t] = acc;
}

// =====================================================================
// Kernel 2: size-pred MLP -> n_logits[b], n[b]   (one block of 128 per row)
// =====================================================================
__global__ void __launch_bounds__(128) sizepred_kernel(
    const float* __restrict__ z,
    const float* __restrict__ sW1, const float* __restrict__ sb1,
    const float* __restrict__ sg,  const float* __restrict__ sbe,
    const float* __restrict__ sW2, const float* __restrict__ sb2,
    float* __restrict__ nlog_out, int* __restrict__ n_out, int* __restrict__ n_ws)
{
    __shared__ float s_z[HIDQ];
    __shared__ float s_red[SMQ];
    const int b = blockIdx.x, t = threadIdx.x;
    s_z[t]        = z[(size_t)b * HIDQ + t];
    s_z[t + SMQ]  = z[(size_t)b * HIDQ + SMQ + t];
    __syncthreads();

    float h = sb1[t];
    #pragma unroll 8
    for (int k = 0; k < HIDQ; ++k) h += s_z[k] * sW1[k * SMQ + t];

    s_red[t] = h; __syncthreads();
    for (int off = 64; off > 0; off >>= 1) { if (t < off) s_red[t] += s_red[t + off]; __syncthreads(); }
    const float mu = s_red[0] * (1.0f / SMQ); __syncthreads();
    const float d = h - mu;
    s_red[t] = d * d; __syncthreads();
    for (int off = 64; off > 0; off >>= 1) { if (t < off) s_red[t] += s_red[t + off]; __syncthreads(); }
    const float var = s_red[0] * (1.0f / SMQ); __syncthreads();

    const float r = fmaxf(d * rsqrtf(var + LN_EPS) * sg[t] + sbe[t], 0.0f);
    s_red[t] = r * sW2[t]; __syncthreads();
    for (int off = 64; off > 0; off >>= 1) { if (t < off) s_red[t] += s_red[t + off]; __syncthreads(); }
    if (t == 0) {
        const float logit = s_red[0] + sb2[0];
        int n = (int)lrintf(logit);
        n = n < 0 ? 0 : (n > MAXNQ - 1 ? MAXNQ - 1 : n);
        nlog_out[b] = logit;
        n_out[b]    = n;
        n_ws[b]     = n;
    }
}

// =====================================================================
// Kernel 3: convert decoder weights to f16, transposed N-major:
//   w1t[n*256 + k] = dW1[k*192 + n]   (192x256)
//   w2t[n*192 + k] = dW2[k*128 + n]   (128x192)
// =====================================================================
__global__ void __launch_bounds__(256) convert_kernel(
    const float* __restrict__ dW1, const float* __restrict__ dW2,
    _Float16* __restrict__ w1t, _Float16* __restrict__ w2t)
{
    const int idx = blockIdx.x * blockDim.x + threadIdx.x;
    if (idx < DMQ * HIDQ) {
        const int n = idx >> 8, k = idx & 255;
        w1t[idx] = (_Float16)dW1[k * DMQ + n];
    }
    if (idx < DIMQ * DMQ) {
        const int n = idx / DMQ, k = idx % DMQ;
        w2t[idx] = (_Float16)dW2[k * DIMQ + n];
    }
}

// =====================================================================
// Kernel 4: fused decoder
//   per (b,j) row: zp = z[b]*keys[j]; x = tanh(zp@dW1+db1)@dW2+db2; mask.
// 512 threads = 16 waves; wave owns a 16-row tile (M=16), 144 WMMAs/tile.
// LDS: w1t 96KB + w2t 48KB + biases + 16 x (16x192) h1 (96KB) ~ 241KB.
// =====================================================================
__global__ void __launch_bounds__(DEC_THREADS) decoder_kernel(
    const float* __restrict__ z, const float* __restrict__ keys,
    const _Float16* __restrict__ w1t, const _Float16* __restrict__ w2t,
    const float* __restrict__ db1, const float* __restrict__ db2,
    const int* __restrict__ n_arr, float* __restrict__ out)
{
    extern __shared__ __align__(16) char smem_raw[];
    _Float16* s_w1t = (_Float16*)smem_raw;                 // [192*256]
    _Float16* s_w2t = s_w1t + DMQ * HIDQ;                  // [128*192]
    float*    s_db1 = (float*)(s_w2t + DIMQ * DMQ);        // [192]
    float*    s_db2 = s_db1 + DMQ;                         // [128]
    _Float16* s_h1  = (_Float16*)(s_db2 + DIMQ);           // [16][16][192]

    const int tid = threadIdx.x;
    // cooperative weight staging (16B chunks)
    {
        const uint4* g1 = (const uint4*)w1t;  uint4* s1 = (uint4*)s_w1t;
        for (int i = tid; i < (DMQ * HIDQ) / 8; i += DEC_THREADS) s1[i] = g1[i];
        const uint4* g2 = (const uint4*)w2t;  uint4* s2 = (uint4*)s_w2t;
        for (int i = tid; i < (DIMQ * DMQ) / 8; i += DEC_THREADS) s2[i] = g2[i];
        if (tid < DMQ)  s_db1[tid] = db1[tid];
        else if (tid >= 256 && tid < 256 + DIMQ) s_db2[tid - 256] = db2[tid - 256];
    }
    __syncthreads();

    const int wave = tid >> 5;
    const int lane = tid & 31;
    const int m    = lane & 15;     // row (A) or column (B/C) within tile
    const int hi   = lane >> 4;     // half-wave select per ISA VGPR layouts

    const size_t row0  = (size_t)blockIdx.x * DEC_ROWS + (size_t)wave * 16;
    const size_t myrow = row0 + (size_t)m;          // this lane's A row
    const float* zrow = z    + (myrow >> 6) * HIDQ;
    const float* krow = keys + (myrow & 63) * HIDQ;

    // ---- build A fragments for GEMM1 in registers: zp = z ⊙ keys -> f16 ----
    v16h afrag[8];
    #pragma unroll
    for (int k8 = 0; k8 < 8; ++k8) {
        const int kb = k8 * 32 + hi * 8;
        v16h a;
        #pragma unroll
        for (int i = 0; i < 8; ++i) a[i]     = (_Float16)(zrow[kb + i]      * krow[kb + i]);
        #pragma unroll
        for (int i = 0; i < 8; ++i) a[8 + i] = (_Float16)(zrow[kb + 16 + i] * krow[kb + 16 + i]);
        afrag[k8] = a;
    }

    // ---- GEMM1: h1 = tanh(zp @ dW1 + db1), 12 N-tiles x 8 K-steps ----
    _Float16* my_h1 = s_h1 + (size_t)wave * (16 * DMQ);
    for (int nt = 0; nt < 12; ++nt) {
        const int nb = nt * 16;
        v8f acc = {};
        const _Float16* bbase = s_w1t + (size_t)(nb + m) * HIDQ + hi * 8;
        #pragma unroll
        for (int k8 = 0; k8 < 8; ++k8) {
            const v16h b = loadB16(bbase + k8 * 32);
            acc = __builtin_amdgcn_wmma_f32_16x16x32_f16(
                      false, afrag[k8], false, b, (short)0, acc, false, false);
        }
        const float bias = s_db1[nb + m];   // C-layout: one column per lane
        #pragma unroll
        for (int v = 0; v < 8; ++v) {
            const int mr = v + hi * 8;      // C row for this VGPR
            my_h1[mr * DMQ + nb + m] = (_Float16)fast_tanh(acc[v] + bias);
        }
    }
    // wave-local LDS RAW fence before re-reading h1 as A-fragments
    asm volatile("s_wait_dscnt 0" ::: "memory");

    // ---- repack h1 into A-fragment layout for GEMM2 (K = 192) ----
    v16h hfrag[6];
    #pragma unroll
    for (int k6 = 0; k6 < 6; ++k6)
        hfrag[k6] = loadA16(my_h1 + m * DMQ + k6 * 32 + hi * 8);

    // per-lane row masks (rows row0 + v + hi*8)
    float maskv[8];
    #pragma unroll
    for (int v = 0; v < 8; ++v) {
        const size_t gr = row0 + (size_t)v + (size_t)hi * 8;
        maskv[v] = ((int)(gr & 63) < n_arr[gr >> 6]) ? 1.0f : 0.0f;
    }

    // ---- GEMM2: x = h1 @ dW2 + db2, 8 N-tiles x 6 K-steps; masked store ----
    for (int nt2 = 0; nt2 < 8; ++nt2) {
        const int nb2 = nt2 * 16;
        v8f acc = {};
        const _Float16* bbase = s_w2t + (size_t)(nb2 + m) * DMQ + hi * 8;
        #pragma unroll
        for (int k6 = 0; k6 < 6; ++k6) {
            const v16h b = loadB16(bbase + k6 * 32);
            acc = __builtin_amdgcn_wmma_f32_16x16x32_f16(
                      false, hfrag[k6], false, b, (short)0, acc, false, false);
        }
        const float bias = s_db2[nb2 + m];
        const int col = nb2 + m;
        #pragma unroll
        for (int v = 0; v < 8; ++v) {
            const size_t gr = row0 + (size_t)v + (size_t)hi * 8;
            out[gr * DIMQ + col] = (acc[v] + bias) * maskv[v];
        }
    }
}

// =====================================================================
// launch
// =====================================================================
extern "C" void kernel_launch(void* const* d_in, const int* in_sizes, int n_in,
                              void* d_out, int out_size, void* d_ws, size_t ws_size,
                              hipStream_t stream) {
    (void)in_sizes; (void)n_in; (void)out_size; (void)ws_size;
    const float* z   = (const float*)d_in[0];
    const float* kW1 = (const float*)d_in[1];
    const float* kb1 = (const float*)d_in[2];
    const float* kg  = (const float*)d_in[3];
    const float* kbe = (const float*)d_in[4];
    const float* kW2 = (const float*)d_in[5];
    const float* kb2 = (const float*)d_in[6];
    const float* dW1 = (const float*)d_in[7];
    const float* db1 = (const float*)d_in[8];
    const float* dW2 = (const float*)d_in[9];
    const float* db2 = (const float*)d_in[10];
    const float* sW1 = (const float*)d_in[11];
    const float* sb1 = (const float*)d_in[12];
    const float* sg  = (const float*)d_in[13];
    const float* sbe = (const float*)d_in[14];
    const float* sW2 = (const float*)d_in[15];
    const float* sb2 = (const float*)d_in[16];

    float* out = (float*)d_out;
    const size_t X_ELEMS = (size_t)BQ * MAXNQ * DIMQ;   // 67,108,864
    float* nlog_out = out + X_ELEMS;                    // [B]
    int*   n_out    = (int*)(out + X_ELEMS + BQ);       // [B] int32 bits

    // workspace layout
    char* ws = (char*)d_ws;
    float*    keys = (float*)ws;                        //  64*256 f32 = 64KB
    int*      n_ws = (int*)(ws + 65536);                //  8192 i32  = 32KB
    _Float16* w1t  = (_Float16*)(ws + 98304);           //  192*256 f16 = 96KB
    _Float16* w2t  = (_Float16*)(ws + 196608);          //  128*192 f16 = 48KB

    keys_kernel<<<MAXNQ, 256, 0, stream>>>(kW1, kb1, kg, kbe, kW2, kb2, keys);
    sizepred_kernel<<<BQ, 128, 0, stream>>>(z, sW1, sb1, sg, sbe, sW2, sb2,
                                            nlog_out, n_out, n_ws);
    convert_kernel<<<(DMQ * HIDQ + 255) / 256, 256, 0, stream>>>(dW1, dW2, w1t, w2t);

    const size_t smem = (size_t)(DMQ * HIDQ + DIMQ * DMQ + DEC_WAVES * 16 * DMQ) * sizeof(_Float16)
                      + (size_t)(DMQ + DIMQ) * sizeof(float);   // ~241 KB (<320 KB WGP LDS)
    const int blocks = (BQ * MAXNQ) / DEC_ROWS;                  // 2048
    decoder_kernel<<<blocks, DEC_THREADS, smem, stream>>>(z, keys, w1t, w2t, db1, db2, n_ws, out);
}
